// GAT_57638461112858
// MI455X (gfx1250) — matrix operand order repaired and checked by reading, not compiled, not run on hardware
//
#include <hip/hip_runtime.h>
#include <hip/hip_bf16.h>

// ---------------------------------------------------------------------------
// GAT (3-layer, PyG-style GATConv) for gfx1250 / MI455X.
//   * Node feature transforms use v_wmma_f32_16x16x32_bf16. One wave per
//     16-row strip holds TN accumulators (TN = Nout/16), so each A fragment
//     is loaded once and feeds TN WMMAs per k-step.
//   * Edge softmax/aggregation uses L2-resident atomics; the message scatter
//     maps one thread per (edge, head, channel) for coalesced gathers and
//     cacheline-grouped global_atomic_add_f32.
// ---------------------------------------------------------------------------

typedef __attribute__((ext_vector_type(16))) __bf16 v16bf;
typedef __attribute__((ext_vector_type(8)))  float  v8f;

#define CDIV(a, b) (((a) + (b) - 1) / (b))

// ---------------------------------------------------------------- utilities
__global__ void k_fill_f32(float* __restrict__ p, float v, long long n) {
    long long i = (long long)blockIdx.x * blockDim.x + threadIdx.x;
    if (i < n) p[i] = v;
}

__global__ void k_f32_to_bf16(const float* __restrict__ s, __bf16* __restrict__ d,
                              long long n) {
    long long i = (long long)blockIdx.x * blockDim.x + threadIdx.x;
    if (i < n) d[i] = (__bf16)s[i];
}

// Pack W[K][Nreal] (f32 row-major) into per-lane WMMA B fragments (bf16).
// Fragment element j of lane l in tile (ks, tn) holds
//   B[k = ks*32 + (l>>4)*8 + (j&7) + (j>>3)*16, n = tn*16 + (l&15)]   (zero-padded)
__global__ void k_pack_w(const float* __restrict__ W, __bf16* __restrict__ Bp,
                         int K, int Nreal, int tilesN, long long total) {
    long long i = (long long)blockIdx.x * blockDim.x + threadIdx.x;
    if (i >= total) return;
    int j    = (int)(i & 15);
    int lane = (int)((i >> 4) & 31);
    long long tnk = i >> 9;
    int tn = (int)(tnk % tilesN);
    int ks = (int)(tnk / tilesN);
    int n = tn * 16 + (lane & 15);
    int k = ks * 32 + (lane >> 4) * 8 + (j & 7) + (j >> 3) * 16;
    float v = (k < K && n < Nreal) ? W[(size_t)k * Nreal + n] : 0.0f;
    Bp[i] = (__bf16)v;
}

// ------------------------------------------------------------- WMMA GEMM
// One wave per 16-row strip; TN accumulators cover the full output width.
// C[M x Nreal] (f32, ldc=Nreal) = A[M x K](bf16, row-major) * B(packed).
template <int TN>
__global__ __launch_bounds__(32) void k_gemm_wmma_bf16(
    const __bf16* __restrict__ A, const __bf16* __restrict__ Bp,
    float* __restrict__ C, int M, int K, int Nreal) {
    int tm   = blockIdx.x;
    int lane = threadIdx.x & 31;
    int l15  = lane & 15;
    int lh   = lane >> 4;

    int m  = tm * 16 + l15;
    int mr = (m < M) ? m : (M - 1);           // clamp; garbage rows never stored
    const __bf16* arow = A + (size_t)mr * K;
    int KT = K >> 5;

    v8f acc[TN];
#pragma unroll
    for (int t = 0; t < TN; ++t) acc[t] = (v8f){};

    for (int ks = 0; ks < KT; ++ks) {
        union { v16bf v; __bf16 e[16]; } af;
        const __bf16* ap = arow + ks * 32 + lh * 8;
#pragma unroll
        for (int q = 0; q < 8; ++q) {         // two contiguous 16B chunks
            af.e[q]     = ap[q];              // K = ks*32 + lh*8 + q
            af.e[8 + q] = ap[16 + q];         // K = ks*32 + 16 + lh*8 + q
        }
        const __bf16* bbase = Bp + (((size_t)ks * TN) * 32 + lane) * 16;
#pragma unroll
        for (int t = 0; t < TN; ++t) {
            v16bf bf = *(const v16bf*)(bbase + (size_t)t * 32 * 16);
            acc[t] = __builtin_amdgcn_wmma_f32_16x16x32_bf16(
                false, af.v, false, bf, (short)0, acc[t], false, false);
        }
    }

#pragma unroll
    for (int t = 0; t < TN; ++t) {
        int n = t * 16 + l15;                 // D layout: lane&15 = column
        if (n < Nreal) {
#pragma unroll
            for (int r = 0; r < 8; ++r) {
                int row = tm * 16 + r + lh * 8;   // lane>>4 selects row half
                if (row < M) C[(size_t)row * Nreal + n] = acc[t][r];
            }
        }
    }
}

// --------------------------------------------------- per-node attn logits
// als[n,h] = <h[n,h,:], a_src[h,:]>,  ald likewise.  One thread per (node,head).
__global__ void k_node_logits(const float* __restrict__ h,
                              const float* __restrict__ as,
                              const float* __restrict__ ad,
                              float* __restrict__ als, float* __restrict__ ald,
                              int Hh, int Cc, long long NH) {
    long long i = (long long)blockIdx.x * blockDim.x + threadIdx.x;
    if (i >= NH) return;
    int hh = (int)(i % Hh);
    long long node = i / Hh;
    const float* hp = h + node * (size_t)(Hh * Cc) + (size_t)hh * Cc;
    const float* ap = as + (size_t)hh * Cc;
    const float* bp = ad + (size_t)hh * Cc;
    float s1 = 0.f, s2 = 0.f;
    for (int c = 0; c < Cc; ++c) {
        float v = hp[c];
        s1 += v * ap[c];
        s2 += v * bp[c];
    }
    als[i] = s1;
    ald[i] = s2;
}

// ---------------------------------------------------------- edge kernels
__device__ __forceinline__ void atomicMaxF(float* addr, float v) {
    if (v >= 0.0f) atomicMax((int*)addr, __float_as_int(v));
    else           atomicMin((unsigned int*)addr, __float_as_uint(v));
}

__device__ __forceinline__ float lrelu(float v) {
    return v > 0.0f ? v : 0.2f * v;
}

__global__ void k_edge_max(const int* __restrict__ src, const int* __restrict__ dst,
                           const float* __restrict__ als, const float* __restrict__ ald,
                           float* __restrict__ mbuf, int Hh, long long EH) {
    long long i = (long long)blockIdx.x * blockDim.x + threadIdx.x;
    if (i >= EH) return;
    int hh = (int)(i % Hh);
    long long e = i / Hh;
    int s = src[e], d = dst[e];
    float v = lrelu(als[(size_t)s * Hh + hh] + ald[(size_t)d * Hh + hh]);
    atomicMaxF(&mbuf[(size_t)d * Hh + hh], v);
}

__global__ void k_edge_expsum(const int* __restrict__ src, const int* __restrict__ dst,
                              const float* __restrict__ als, const float* __restrict__ ald,
                              const float* __restrict__ mbuf,
                              float* __restrict__ exb, float* __restrict__ dbuf,
                              int Hh, long long EH) {
    long long i = (long long)blockIdx.x * blockDim.x + threadIdx.x;
    if (i >= EH) return;
    int hh = (int)(i % Hh);
    long long e = i / Hh;
    int s = src[e], d = dst[e];
    float v  = lrelu(als[(size_t)s * Hh + hh] + ald[(size_t)d * Hh + hh]);
    float ex = __expf(v - mbuf[(size_t)d * Hh + hh]);
    exb[i] = ex;
    atomicAdd(&dbuf[(size_t)d * Hh + hh], ex);
}

// One thread per (edge, head, channel): 32 consecutive lanes touch one
// contiguous channel block of h[src] / obuf[dst] -> coalesced loads and
// cacheline-grouped f32 atomics.
__global__ void k_edge_msg(const int* __restrict__ src, const int* __restrict__ dst,
                           const float* __restrict__ exb, const float* __restrict__ dbuf,
                           const float* __restrict__ h, float* __restrict__ obuf,
                           int Hh, int Cc, long long EHC) {
    long long i = (long long)blockIdx.x * blockDim.x + threadIdx.x;
    if (i >= EHC) return;
    int c = (int)(i % Cc);
    long long t = i / Cc;                 // t = e*Hh + hh
    int hh = (int)(t % Hh);
    long long e = t / Hh;
    int s = src[e], d = dst[e];
    float alpha = exb[t] / dbuf[(size_t)d * Hh + hh];
    float v = h[(size_t)s * (Hh * Cc) + (size_t)hh * Cc + c] * alpha;
    atomicAdd(&obuf[(size_t)d * (Hh * Cc) + (size_t)hh * Cc + c], v);
}

// -------------------------------------------------- epilogues per layer
__global__ void k_bias_elu_to_bf16(const float* __restrict__ o,
                                   const float* __restrict__ b,
                                   __bf16* __restrict__ xa, int HC, long long n) {
    long long i = (long long)blockIdx.x * blockDim.x + threadIdx.x;
    if (i >= n) return;
    int col = (int)(i % HC);
    float v = o[i] + b[col];
    v = v > 0.0f ? v : (__expf(v) - 1.0f);     // ELU(alpha=1)
    xa[i] = (__bf16)v;
}

__global__ void k_bias_logsoftmax(const float* __restrict__ o,
                                  const float* __restrict__ b,
                                  float* __restrict__ out, int Cc, long long Nn) {
    long long node = (long long)blockIdx.x * blockDim.x + threadIdx.x;
    if (node >= Nn) return;
    const float* op = o + node * (size_t)Cc;
    float mx = -3.402823466e38f;
    for (int c = 0; c < Cc; ++c) {
        float v = op[c] + b[c];
        mx = v > mx ? v : mx;
    }
    float sum = 0.f;
    for (int c = 0; c < Cc; ++c)
        sum += __expf(op[c] + b[c] - mx);
    float lse = mx + __logf(sum);
    for (int c = 0; c < Cc; ++c)
        out[node * (size_t)Cc + c] = op[c] + b[c] - lse;
}

// ---------------------------------------------------------------------------
extern "C" void kernel_launch(void* const* d_in, const int* in_sizes, int n_in,
                              void* d_out, int out_size, void* d_ws, size_t ws_size,
                              hipStream_t stream) {
    const float* x    = (const float*)d_in[0];
    const int*   ei   = (const int*)d_in[1];
    const float* W1   = (const float*)d_in[2];
    const float* as1  = (const float*)d_in[3];
    const float* ad1  = (const float*)d_in[4];
    const float* b1   = (const float*)d_in[5];
    const float* W2   = (const float*)d_in[6];
    const float* as2  = (const float*)d_in[7];
    const float* ad2  = (const float*)d_in[8];
    const float* b2   = (const float*)d_in[9];
    const float* W3   = (const float*)d_in[10];
    const float* as3  = (const float*)d_in[11];
    const float* ad3  = (const float*)d_in[12];
    const float* b3   = (const float*)d_in[13];

    const long long N = in_sizes[0] / 128;    // nodes
    const long long E = in_sizes[1] / 2;      // edges (incl. self-loops)
    const int* src = ei;
    const int* dst = ei + E;

    // ---- carve workspace --------------------------------------------------
    size_t off = 0;
    auto carve = [&](size_t bytes) -> void* {
        void* p = (char*)d_ws + off;
        off += (bytes + 255) & ~(size_t)255;
        return p;
    };
    __bf16* xa   = (__bf16*)carve((size_t)N * 128 * sizeof(__bf16)); // GEMM A (bf16)
    float*  hbuf = (float*)carve((size_t)N * 64 * sizeof(float));    // GEMM out h
    float*  obuf = (float*)carve((size_t)N * 64 * sizeof(float));    // aggregation
    float*  als  = (float*)carve((size_t)N * 2 * sizeof(float));
    float*  ald  = (float*)carve((size_t)N * 2 * sizeof(float));
    float*  mbuf = (float*)carve((size_t)N * 2 * sizeof(float));
    float*  dbuf = (float*)carve((size_t)N * 2 * sizeof(float));
    float*  exb  = (float*)carve((size_t)E * 2 * sizeof(float));
    __bf16* wpk  = (__bf16*)carve(65536);                            // packed W tiles
    (void)ws_size; (void)n_in; (void)out_size;

    const int TB = 256;
    const int tilesM = (int)CDIV(N, 16);
    const float NEG_INF = -3.402823466e38f;

    // x -> bf16 once
    {
        long long n = N * 128;
        k_f32_to_bf16<<<CDIV(n, TB), TB, 0, stream>>>(x, xa, n);
    }

    // per-layer runner -------------------------------------------------------
    auto run_layer = [&](const float* W, const float* as, const float* ad,
                         int K, int Nreal, int Hh, int Cc) {
        const int tilesN = CDIV(Nreal, 16);
        const int KT = K / 32;
        // pack W into WMMA B fragments
        long long pk = (long long)KT * tilesN * 512;
        k_pack_w<<<CDIV(pk, TB), TB, 0, stream>>>(W, wpk, K, Nreal, tilesN, pk);
        // h = xa * W  (WMMA): one wave per 16-row strip, TN accumulators
        if (tilesN == 4) {
            k_gemm_wmma_bf16<4><<<tilesM, 32, 0, stream>>>(xa, wpk, hbuf, (int)N, K, Nreal);
        } else {
            k_gemm_wmma_bf16<3><<<tilesM, 32, 0, stream>>>(xa, wpk, hbuf, (int)N, K, Nreal);
        }
        // attention logits
        long long NH = N * Hh;
        k_node_logits<<<CDIV(NH, TB), TB, 0, stream>>>(hbuf, as, ad, als, ald, Hh, Cc, NH);
        // init accumulators
        k_fill_f32<<<CDIV(NH, TB), TB, 0, stream>>>(mbuf, NEG_INF, NH);
        k_fill_f32<<<CDIV(NH, TB), TB, 0, stream>>>(dbuf, 0.0f, NH);
        long long no = N * (long long)(Hh * Cc);
        k_fill_f32<<<CDIV(no, TB), TB, 0, stream>>>(obuf, 0.0f, no);
        // edge phase
        long long EH = E * Hh;
        k_edge_max   <<<CDIV(EH, TB), TB, 0, stream>>>(src, dst, als, ald, mbuf, Hh, EH);
        k_edge_expsum<<<CDIV(EH, TB), TB, 0, stream>>>(src, dst, als, ald, mbuf, exb, dbuf, Hh, EH);
        long long EHC = EH * Cc;
        k_edge_msg   <<<CDIV(EHC, TB), TB, 0, stream>>>(src, dst, exb, dbuf, hbuf, obuf, Hh, Cc, EHC);
    };

    // ---- layer 1: 128 -> 2x32 (concat 64), ELU ----------------------------
    run_layer(W1, as1, ad1, /*K=*/128, /*Nreal=*/64, /*Hh=*/2, /*Cc=*/32);
    {
        long long n = N * 64;
        k_bias_elu_to_bf16<<<CDIV(n, TB), TB, 0, stream>>>(obuf, b1, xa, 64, n);
    }

    // ---- layer 2: 64 -> 2x32 (concat 64), ELU -----------------------------
    run_layer(W2, as2, ad2, /*K=*/64, /*Nreal=*/64, /*Hh=*/2, /*Cc=*/32);
    {
        long long n = N * 64;
        k_bias_elu_to_bf16<<<CDIV(n, TB), TB, 0, stream>>>(obuf, b2, xa, 64, n);
    }

    // ---- layer 3: 64 -> 1x40 (no concat), log_softmax ---------------------
    run_layer(W3, as3, ad3, /*K=*/64, /*Nreal=*/40, /*Hh=*/1, /*Cc=*/40);
    k_bias_logsoftmax<<<CDIV(N, TB), TB, 0, stream>>>(obuf, b3, (float*)d_out, 40, N);
}
